// MultiAttentionAggr_75909251990140
// MI455X (gfx1250) — matrix operand order repaired and checked by reading, not compiled, not run on hardware
//
#include <hip/hip_runtime.h>

#define N_NODES    100000
#define DIM        256
#define HID        128
#define HEADS      4
#define NUM_GRAPHS 1024

// ---- CDNA5 WMMA vector types (wave32, 16x16x32 bf16 -> f32) ----
typedef __attribute__((ext_vector_type(16))) __bf16       v16bf;
typedef __attribute__((ext_vector_type(2)))  __bf16       v2bf;
typedef __attribute__((ext_vector_type(8)))  float        v8f;
typedef __attribute__((ext_vector_type(8)))  unsigned int v8u;

// Native 4xi32 vector (what the async-to-LDS builtin expects), with
// address-space-qualified pointer typedefs (AS1 = global, AS3 = LDS).
typedef int nint4 __attribute__((vector_size(4 * sizeof(int))));
typedef __attribute__((address_space(1))) nint4 g_nint4;
typedef __attribute__((address_space(3))) nint4 l_nint4;

#if __has_builtin(__builtin_amdgcn_global_load_async_to_lds_b128)
#define HAVE_ASYNC_LDS 1
#else
#define HAVE_ASYNC_LDS 0
#endif

// fp32 -> bf16 pair via native fptrunc (lowers to v_cvt_{pk_}bf16_f32 on
// gfx1250, round-to-nearest-even); lo in bits [15:0], hi in [31:16].
__device__ __forceinline__ unsigned int pack_bf16x2(float lo, float hi) {
  v2bf p;
  p[0] = (__bf16)lo;
  p[1] = (__bf16)hi;
  return __builtin_bit_cast(unsigned int, p);
}

// Build an A fragment (16x32 bf16, this lane's 16 K-values) from an fp32 row.
__device__ __forceinline__ v16bf make_a_frag(const float* __restrict__ ap) {
  float4 c0 = *reinterpret_cast<const float4*>(ap);
  float4 c1 = *reinterpret_cast<const float4*>(ap + 4);
  float4 c2 = *reinterpret_cast<const float4*>(ap + 16);
  float4 c3 = *reinterpret_cast<const float4*>(ap + 20);
  v16bf a;
  a[0]  = (__bf16)c0.x; a[1]  = (__bf16)c0.y;
  a[2]  = (__bf16)c0.z; a[3]  = (__bf16)c0.w;
  a[4]  = (__bf16)c1.x; a[5]  = (__bf16)c1.y;
  a[6]  = (__bf16)c1.z; a[7]  = (__bf16)c1.w;
  a[8]  = (__bf16)c2.x; a[9]  = (__bf16)c2.y;
  a[10] = (__bf16)c2.z; a[11] = (__bf16)c2.w;
  a[12] = (__bf16)c3.x; a[13] = (__bf16)c3.y;
  a[14] = (__bf16)c3.z; a[15] = (__bf16)c3.w;
  return a;
}

// Load a pre-swizzled B fragment (32x16 bf16) from LDS: 32B contiguous/lane.
__device__ __forceinline__ v16bf load_b_frag(const unsigned int* lds, int ks,
                                             int ct, int lane) {
  const int4* bp = reinterpret_cast<const int4*>(
      lds + (((ks * 8 + ct) * 32 + lane) << 3));
  int4 bA = bp[0];
  int4 bB = bp[1];
  v8u bu;
  bu[0] = (unsigned)bA.x; bu[1] = (unsigned)bA.y;
  bu[2] = (unsigned)bA.z; bu[3] = (unsigned)bA.w;
  bu[4] = (unsigned)bB.x; bu[5] = (unsigned)bB.y;
  bu[6] = (unsigned)bB.z; bu[7] = (unsigned)bB.w;
  return __builtin_bit_cast(v16bf, bu);
}

// -------------------------------------------------------------------------
// Kernel 1: pre-swizzle W1 (fp32 [H,DIM,HID]) into bf16 B-fragment layout.
// Pack layout: [h][ks(8)][ct(8)][lane(32)][j(8)] dwords; lane holds column
// k = 16*ct + (lane&15); its 16 K-values (d-dim) are d = 32*ks + 8*(lane>>4)
// + {0..7} then 32*ks + 16 + 8*(lane>>4) + {0..7}, two bf16 per dword. Lane's 8 dwords are contiguous (32B).
// -------------------------------------------------------------------------
__global__ void pack_w1_kernel(const float* __restrict__ W1,
                               unsigned int* __restrict__ packW1) {
  int idx = blockIdx.x * blockDim.x + threadIdx.x;  // 65536 total
  int j  = idx & 7;
  int L  = (idx >> 3) & 31;
  int ct = (idx >> 8) & 7;
  int ks = (idx >> 11) & 7;
  int h  = (idx >> 14) & 3;
  int hl = L >> 4;
  int d0 = ks * 32 + (j >> 2) * 16 + hl * 8 + 2 * (j & 3);
  int k  = ct * 16 + (L & 15);
  float w0 = W1[((size_t)(h * DIM) + d0) * HID + k];
  float w1 = W1[((size_t)(h * DIM) + d0 + 1) * HID + k];
  packW1[idx] = pack_bf16x2(w0, w1);
}

// -------------------------------------------------------------------------
// Kernel 2: segment boundaries from sorted batch. starts[g] = first node of
// graph g; starts[NUM_GRAPHS] = N. Empty graphs get start == end.
// -------------------------------------------------------------------------
__global__ void seg_starts_kernel(const int* __restrict__ batch,
                                  int* __restrict__ starts) {
  int n = blockIdx.x * blockDim.x + threadIdx.x;
  if (n >= N_NODES) return;
  int cur  = batch[n];
  int prev = (n == 0) ? -1 : batch[n - 1];
  for (int g = prev + 1; g <= cur; ++g) starts[g] = n;
  if (n == N_NODES - 1)
    for (int g = cur + 1; g <= NUM_GRAPHS; ++g) starts[g] = N_NODES;
}

// -------------------------------------------------------------------------
// Kernel 3: gate[h][n] = relu(x[n,:] @ W1[h] + b1[h]) . W2[h]
// grid = (ceil(N/256), HEADS), block = 256 (8 waves). Wave w owns a 32-node
// tile (two 16-row A fragments) so each B fragment feeds two WMMAs; B is
// double-buffered across the ct loop so ds_load overlaps v_wmma.
// W1 pack for the head is staged into LDS via GLOBAL_LOAD_ASYNC_TO_LDS_B128
// (ASYNCcnt path), else a plain LDS copy if the builtin is absent.
// (b2 is uniform per head -> cancels in segment softmax, omitted.)
// -------------------------------------------------------------------------
__global__ void __launch_bounds__(256)
gate_gemm_kernel(const float* __restrict__ x,
                 const unsigned int* __restrict__ packW1,
                 const float* __restrict__ b1,
                 const float* __restrict__ W2,
                 float* __restrict__ gate) {
  __shared__ unsigned int ldsW[16384];  // 64 KB: one head's packed W1

  const int h = blockIdx.y;
  {
    const int4* src = reinterpret_cast<const int4*>(packW1 + h * 16384);
    int4* dst = reinterpret_cast<int4*>(ldsW);
#if HAVE_ASYNC_LDS
    for (int i = threadIdx.x; i < 4096; i += 256) {
      g_nint4* gp = (g_nint4*)(uintptr_t)(src + i);
      l_nint4* lp = (l_nint4*)(unsigned int)(uintptr_t)(dst + i);
      __builtin_amdgcn_global_load_async_to_lds_b128(gp, lp, 0, 0);
    }
#if __has_builtin(__builtin_amdgcn_s_wait_asynccnt)
    __builtin_amdgcn_s_wait_asynccnt(0);
#else
    asm volatile("s_wait_asynccnt 0x0" ::: "memory");
#endif
#else
    for (int i = threadIdx.x; i < 4096; i += 256) dst[i] = src[i];
#endif
  }
  __syncthreads();

  const int wave = threadIdx.x >> 5;
  const int lane = threadIdx.x & 31;
  const int kh   = lane >> 4;   // lane half (K-group select)
  const int l15  = lane & 15;   // A row / D column within tile
  const int nodeBase = blockIdx.x * 256 + wave * 32;

  int r0 = nodeBase + l15;
  int r1 = nodeBase + 16 + l15;
  if (r0 >= N_NODES) r0 = N_NODES - 1;  // clamp; gate writes are guarded
  if (r1 >= N_NODES) r1 = N_NODES - 1;
  const float* __restrict__ xr0 = x + (size_t)r0 * DIM;
  const float* __restrict__ xr1 = x + (size_t)r1 * DIM;

  v8f acc0[8] = {};  // node rows 0..15 of the tile, 8 column tiles
  v8f acc1[8] = {};  // node rows 16..31

#pragma unroll
  for (int ks = 0; ks < 8; ++ks) {
    v16bf a0 = make_a_frag(xr0 + ks * 32 + kh * 8);
    v16bf a1 = make_a_frag(xr1 + ks * 32 + kh * 8);
    v16bf bcur = load_b_frag(ldsW, ks, 0, lane);
#pragma unroll
    for (int ct = 0; ct < 8; ++ct) {
      v16bf bnext = bcur;
      if (ct < 7) bnext = load_b_frag(ldsW, ks, ct + 1, lane);
      acc0[ct] = __builtin_amdgcn_wmma_f32_16x16x32_bf16(
          false, a0, false, bcur, (short)0, acc0[ct], false, false);
      acc1[ct] = __builtin_amdgcn_wmma_f32_16x16x32_bf16(
          false, a1, false, bcur, (short)0, acc1[ct], false, false);
      bcur = bnext;
    }
  }

  // Epilogue: relu(+b1) then dot with W2 over the 128 hidden columns.
  // accX[ct][j] holds element (row = j + 8*kh, col = 16*ct + l15).
  float part0[8] = {0.f, 0.f, 0.f, 0.f, 0.f, 0.f, 0.f, 0.f};
  float part1[8] = {0.f, 0.f, 0.f, 0.f, 0.f, 0.f, 0.f, 0.f};
#pragma unroll
  for (int ct = 0; ct < 8; ++ct) {
    int k = ct * 16 + l15;
    float w2c = W2[h * HID + k];
    float b1c = b1[h * HID + k];
#pragma unroll
    for (int j = 0; j < 8; ++j) {
      float v0 = acc0[ct][j] + b1c;
      float v1 = acc1[ct][j] + b1c;
      v0 = v0 > 0.f ? v0 : 0.f;
      v1 = v1 > 0.f ? v1 : 0.f;
      part0[j] += v0 * w2c;
      part1[j] += v1 * w2c;
    }
  }
  // reduce across the 16 lanes of this half (same rows): xor masks < 16
#pragma unroll
  for (int j = 0; j < 8; ++j) {
#pragma unroll
    for (int off = 8; off >= 1; off >>= 1) {
      part0[j] += __shfl_xor(part0[j], off, 32);
      part1[j] += __shfl_xor(part1[j], off, 32);
    }
  }
  if (l15 == 0) {
#pragma unroll
    for (int j = 0; j < 8; ++j) {
      int n0 = nodeBase + kh * 8 + j;
      int n1 = nodeBase + 16 + kh * 8 + j;
      if (n0 < N_NODES) gate[(size_t)h * N_NODES + n0] = part0[j];
      if (n1 < N_NODES) gate[(size_t)h * N_NODES + n1] = part1[j];
    }
  }
}

// -------------------------------------------------------------------------
// Kernel 4: per-graph segment softmax + alpha-weighted feature sum.
// One block per graph, 256 threads = one feature dim each, 4 head accums.
// -------------------------------------------------------------------------
__global__ void __launch_bounds__(256)
pool_kernel(const float* __restrict__ x, const float* __restrict__ gate,
            const int* __restrict__ starts, float* __restrict__ out) {
  __shared__ float sred[256];
  __shared__ float salpha[256];  // [4 heads][64 nodes]

  const int g = blockIdx.x;
  const int t = threadIdx.x;
  const int s = starts[g];
  const int e = starts[g + 1];

  const float NEG_INF = -__builtin_inff();
  float mx[4] = {NEG_INF, NEG_INF, NEG_INF, NEG_INF};
  for (int n = s + t; n < e; n += 256) {
#pragma unroll
    for (int h = 0; h < 4; ++h)
      mx[h] = fmaxf(mx[h], gate[(size_t)h * N_NODES + n]);
  }
#pragma unroll
  for (int h = 0; h < 4; ++h) {
    sred[t] = mx[h];
    __syncthreads();
    for (int st = 128; st > 0; st >>= 1) {
      if (t < st) sred[t] = fmaxf(sred[t], sred[t + st]);
      __syncthreads();
    }
    mx[h] = sred[0];
    __syncthreads();
  }

  float sm[4] = {0.f, 0.f, 0.f, 0.f};
  for (int n = s + t; n < e; n += 256) {
#pragma unroll
    for (int h = 0; h < 4; ++h)
      sm[h] += __expf(gate[(size_t)h * N_NODES + n] - mx[h]);
  }
#pragma unroll
  for (int h = 0; h < 4; ++h) {
    sred[t] = sm[h];
    __syncthreads();
    for (int st = 128; st > 0; st >>= 1) {
      if (t < st) sred[t] += sred[t + st];
      __syncthreads();
    }
    sm[h] = sred[0];
    __syncthreads();
  }
  float rden[4];
#pragma unroll
  for (int h = 0; h < 4; ++h) rden[h] = (sm[h] > 0.f) ? 1.f / sm[h] : 0.f;

  float accv[4] = {0.f, 0.f, 0.f, 0.f};
  for (int base = s; base < e; base += 64) {
    int cnt = e - base;
    if (cnt > 64) cnt = 64;
    __syncthreads();
    {
      int hh = t >> 6, ii = t & 63;
      salpha[t] = (ii < cnt)
          ? __expf(gate[(size_t)hh * N_NODES + base + ii] - mx[hh]) * rden[hh]
          : 0.f;
    }
    __syncthreads();
    for (int i = 0; i < cnt; ++i) {
      float xv = x[(size_t)(base + i) * DIM + t];
#pragma unroll
      for (int h = 0; h < 4; ++h) accv[h] += salpha[h * 64 + i] * xv;
    }
  }
#pragma unroll
  for (int h = 0; h < 4; ++h)
    out[(size_t)g * (HEADS * DIM) + h * DIM + t] = accv[h];
}

// -------------------------------------------------------------------------
extern "C" void kernel_launch(void* const* d_in, const int* in_sizes, int n_in,
                              void* d_out, int out_size, void* d_ws,
                              size_t ws_size, hipStream_t stream) {
  const float* x   = (const float*)d_in[0];
  const int* batch = (const int*)d_in[1];
  const float* W1  = (const float*)d_in[2];
  const float* b1  = (const float*)d_in[3];
  const float* W2  = (const float*)d_in[4];
  // d_in[5] = b2: uniform per head -> cancels in segment softmax; unused.

  float* out = (float*)d_out;

  unsigned char* ws = (unsigned char*)d_ws;
  unsigned int* packW1 = (unsigned int*)ws;                       // 256 KB
  float* gate = (float*)(ws + (size_t)65536 * 4);                 // 1.6 MB
  int* starts = (int*)(ws + (size_t)65536 * 4 +
                       (size_t)HEADS * N_NODES * 4);              // 4.1 KB

  pack_w1_kernel<<<256, 256, 0, stream>>>(W1, packW1);
  seg_starts_kernel<<<(N_NODES + 255) / 256, 256, 0, stream>>>(batch, starts);
  gate_gemm_kernel<<<dim3((N_NODES + 255) / 256, HEADS), 256, 0, stream>>>(
      x, packW1, b1, W2, gate);
  pool_kernel<<<NUM_GRAPHS, 256, 0, stream>>>(x, gate, starts, out);
}
